// SparseAttentionSYCL_39874476376194
// MI455X (gfx1250) — matrix operand order repaired and checked by reading, not compiled, not run on hardware
//
#include <hip/hip_runtime.h>
#include <hip/hip_bf16.h>
#include <math.h>

// ---------------- problem constants (from reference) ----------------
#define HN 16      // heads
#define LN 2048    // sequence length
#define DN 64      // head dim
#define MB 32      // query blocks (L/64)
#define NB 32      // key blocks
#define TOPK 6     // int(0.2 * 32)
#define LDB 72     // bf16 LDS row stride: 144B rows -> 16B aligned, odd word count

typedef __attribute__((ext_vector_type(16))) __bf16 bf16x16;
typedef __attribute__((ext_vector_type(8)))  __bf16 bf16x8;
typedef __attribute__((ext_vector_type(4)))  __bf16 bf16x4;
typedef __attribute__((ext_vector_type(8)))  float  f32x8;

__device__ __forceinline__ f32x8 wmma_bf16(bf16x16 a, bf16x16 b, f32x8 c) {
  return __builtin_amdgcn_wmma_f32_16x16x32_bf16(false, a, false, b, (short)0, c,
                                                 false, false);
}

// A fragment (16x32, row m per lane; elems 0..7 -> K=hi*8+i, 8..15 -> K=16+hi*8+i)
// Source row-major [m][k], runs contiguous -> 2x ds_load_b128.
__device__ __forceinline__ bf16x16 frag_a_w(const __bf16* s, int r0, int c0, int lane) {
  int m = lane & 15, hi = (lane >> 4) & 1;
  const __bf16* base = s + (r0 + m) * LDB + c0 + hi * 8;
  bf16x8 lo = *(const bf16x8*)(base);
  bf16x8 hh = *(const bf16x8*)(base + 16);
  return __builtin_shufflevector(lo, hh, 0, 1, 2, 3, 4, 5, 6, 7, 8, 9, 10, 11, 12,
                                 13, 14, 15);
}
// B fragment (32x16, column n per lane; elems 0..7 -> K=hi*8+i, 8..15 -> K=16+hi*8+i), where B[k][n] is
// stored TRANSPOSED: s[(c0+n)*LDB + r0 + k] -> two contiguous runs of 8 bf16 per lane.
__device__ __forceinline__ bf16x16 frag_bt_w(const __bf16* s, int r0, int c0, int lane) {
  int n = lane & 15, hi = (lane >> 4) & 1;
  const __bf16* base = s + (c0 + n) * LDB + r0 + hi * 8;
  bf16x8 lo = *(const bf16x8*)(base);
  bf16x8 hh = *(const bf16x8*)(base + 16);
  return __builtin_shufflevector(lo, hh, 0, 1, 2, 3, 4, 5, 6, 7, 8, 9, 10, 11, 12,
                                 13, 14, 15);
}

// stage a 64x64 f32 global tile into row-major bf16 LDS (float4 loads, b64 stores)
__device__ __forceinline__ void stage_rowmajor(const float* __restrict__ g,
                                               __bf16* s, int tid, float mul) {
  const float4* g4 = (const float4*)g;
  for (int e = tid; e < 1024; e += 256) {
    float4 x = g4[e];
    int r = e >> 4, c = (e & 15) * 4;
    bf16x4 b;
    b[0] = (__bf16)(x.x * mul);
    b[1] = (__bf16)(x.y * mul);
    b[2] = (__bf16)(x.z * mul);
    b[3] = (__bf16)(x.w * mul);
    *(bf16x4*)(s + r * LDB + c) = b;
  }
}
// stage a 64x64 f32 global tile TRANSPOSED into bf16 LDS (coalesced reads, b16 scatter)
__device__ __forceinline__ void stage_transposed(const float* __restrict__ g,
                                                 __bf16* s, int tid) {
  const float4* g4 = (const float4*)g;
  for (int e = tid; e < 1024; e += 256) {
    float4 x = g4[e];
    int r = e >> 4, c = (e & 15) * 4;
    s[(c + 0) * LDB + r] = (__bf16)x.x;
    s[(c + 1) * LDB + r] = (__bf16)x.y;
    s[(c + 2) * LDB + r] = (__bf16)x.z;
    s[(c + 3) * LDB + r] = (__bf16)x.w;
  }
}

// ---------------- kernel 1: block mean-pool of q and k ----------------
__global__ void means_kernel(const float* __restrict__ q, const float* __restrict__ k,
                             float* __restrict__ qm, float* __restrict__ km) {
  int hb = blockIdx.x;
  int h = hb >> 5, blk = hb & 31;
  int t = threadIdx.x;  // 0..127
  const float* src = (t < 64) ? q : k;
  float* dst = (t < 64) ? qm : km;
  int d = t & 63;
  const float* base = src + ((size_t)(h * LN) + blk * 64) * DN + d;
  float s = 0.f;
#pragma unroll 8
  for (int r = 0; r < 64; ++r) s += base[r * DN];
  dst[(h * MB + blk) * DN + d] = s * (1.0f / 64.0f);
}

// ---------------- kernel 2: top-k block LUT (one wave32 per (h,m)) ----------------
__global__ void topk_kernel(const float* __restrict__ qm, const float* __restrict__ km,
                            int* __restrict__ lut) {
  int hb = blockIdx.x;
  int h = hb >> 5, m = hb & 31;
  int lane = threadIdx.x;  // lane == key-block index n
  const float* qrow = qm + (h * MB + m) * DN;
  const float* krow = km + (h * NB + lane) * DN;
  float s = 0.f;
#pragma unroll 8
  for (int d = 0; d < DN; ++d) s += qrow[d] * krow[d];
  bool taken = false;
  for (int t = 0; t < TOPK; ++t) {
    float bs = taken ? -INFINITY : s;
    int bi = lane;
    for (int off = 16; off > 0; off >>= 1) {
      float os = __shfl_xor(bs, off, 32);
      int   oi = __shfl_xor(bi, off, 32);
      if (os > bs || (os == bs && oi < bi)) { bs = os; bi = oi; }
    }
    if (lane == 0) lut[(h * MB + m) * 8 + t] = bi;
    if (lane == bi) taken = true;  // lax.top_k lower-index tie-break
  }
}

// ---------------- kernel 3: per-head kv_sum = k_feat^T v, k_sum ----------------
__global__ void __launch_bounds__(256)
kv_kernel(const float* __restrict__ k, const float* __restrict__ v,
          float* __restrict__ kvsum, float* __restrict__ ksum) {
  __shared__ __bf16 KFt[64 * LDB];  // k_feat transposed: KFt[d][l]
  __shared__ __bf16 Vt[64 * LDB];   // v transposed:      Vt[e][l]
  int h = blockIdx.x;
  int tid = threadIdx.x, lane = tid & 31, wid = tid >> 5;
  int t0 = wid * 2;
  f32x8 A0 = {}, A1 = {};
  float kacc = 0.f;
  for (int c = 0; c < LN / 64; ++c) {
    const float* kbase = k + ((size_t)(h * LN) + c * 64) * DN;
    const float* vbase = v + ((size_t)(h * LN) + c * 64) * DN;
    stage_transposed(vbase, Vt, tid);
    if (tid < 64) {  // row softmax of k chunk, written transposed as bf16
      int r = tid;
      const float* row = kbase + r * 64;
      float mx = -INFINITY;
      for (int d = 0; d < 64; ++d) mx = fmaxf(mx, row[d]);
      float sum = 0.f;
      for (int d = 0; d < 64; ++d) sum += __expf(row[d] - mx);
      float inv = 1.0f / sum;
      for (int d = 0; d < 64; ++d)
        KFt[d * LDB + r] = (__bf16)(__expf(row[d] - mx) * inv);
    }
    __syncthreads();
    if (tid < 64) {  // k_sum: row of KFt is contiguous
      float cs = 0.f;
      for (int l = 0; l < 64; ++l) cs += (float)KFt[tid * LDB + l];
      kacc += cs;
    }
    for (int tt = 0; tt < 2; ++tt) {  // kv_sum[d][e] += kf^T @ v
      int t = t0 + tt, ti = t >> 2, tj = t & 3;
      f32x8& A = tt ? A1 : A0;
      for (int kk = 0; kk < 64; kk += 32) {
        bf16x16 a = frag_a_w(KFt, ti * 16, kk, lane);  // A[d][l], row-major in KFt
        bf16x16 b = frag_bt_w(Vt, kk, tj * 16, lane);  // B[l][e] = Vt[e][l]
        A = wmma_bf16(a, b, A);
      }
    }
    __syncthreads();
  }
  int hi = (lane >> 4) & 1, nn = lane & 15;
  for (int tt = 0; tt < 2; ++tt) {
    int t = t0 + tt, ti = t >> 2, tj = t & 3;
    f32x8& A = tt ? A1 : A0;
#pragma unroll
    for (int r = 0; r < 8; ++r)
      kvsum[((size_t)h * 64 + ti * 16 + hi * 8 + r) * 64 + tj * 16 + nn] = A[r];
  }
  if (tid < 64) ksum[h * 64 + tid] = kacc;
}

// ---------------- kernel 4: block-sparse flash attention ----------------
__global__ void __launch_bounds__(256)
attn_kernel(const float* __restrict__ q, const float* __restrict__ k,
            const float* __restrict__ v, const int* __restrict__ lut,
            float* __restrict__ out) {
  __shared__ __bf16 Qs[64 * LDB];  // Q*scale, row-major [m][d]
  __shared__ __bf16 Ks[64 * LDB];  // K, row-major [key][d] (B=K^T reads rows)
  __shared__ __bf16 Vt[64 * LDB];  // V transposed [d][key]
  __shared__ __bf16 Pb[64 * LDB];  // P bf16, row-major [m][key]
  __shared__ float  Sf[64 * 65];   // raw scores f32
  __shared__ float  mi[64], li[64], al[64];
  int hb = blockIdx.x;
  int h = hb >> 5, mblk = hb & 31;
  int tid = threadIdx.x, lane = tid & 31, wid = tid >> 5;
  const float* qbase = q + ((size_t)(h * LN) + mblk * 64) * DN;
  stage_rowmajor(qbase, Qs, tid, 0.125f);  // fold scale = 1/sqrt(64) into Q
  if (tid < 64) { mi[tid] = -INFINITY; li[tid] = 0.f; }
  int t0 = wid * 2;
  f32x8 O0 = {}, O1 = {};
  __syncthreads();

  for (int it = 0; it < TOPK; ++it) {
    int nblk = lut[(h * MB + mblk) * 8 + it];
    const float* kbase = k + ((size_t)(h * LN) + nblk * 64) * DN;
    const float* vbase = v + ((size_t)(h * LN) + nblk * 64) * DN;
    stage_rowmajor(kbase, Ks, tid, 1.0f);
    stage_transposed(vbase, Vt, tid);
    __syncthreads();

    // S = Q @ K^T : all operand fragments are wide LDS loads
    for (int tt = 0; tt < 2; ++tt) {
      int t = t0 + tt, ti = t >> 2, tj = t & 3;
      f32x8 S = {};
      for (int kk = 0; kk < 64; kk += 32) {
        bf16x16 a = frag_a_w(Qs, ti * 16, kk, lane);
        bf16x16 b = frag_bt_w(Ks, kk, tj * 16, lane);  // B[d][key] = Ks[key][d]
        S = wmma_bf16(a, b, S);
      }
      int hi = (lane >> 4) & 1, nn = lane & 15;
#pragma unroll
      for (int r = 0; r < 8; ++r)
        Sf[(ti * 16 + hi * 8 + r) * 65 + tj * 16 + nn] = S[r];
    }
    __syncthreads();

    // online softmax row stats; P written once as bf16 (row-major)
    if (tid < 64) {
      int r = tid;
      float rowmax = -INFINITY;
      for (int c = 0; c < 64; ++c) rowmax = fmaxf(rowmax, Sf[r * 65 + c]);
      float newm = fmaxf(mi[r], rowmax);
      float a = __expf(mi[r] - newm);
      float ssum = 0.f;
      for (int c = 0; c < 64; ++c) {
        float p = __expf(Sf[r * 65 + c] - newm);
        Pb[r * LDB + c] = (__bf16)p;
        ssum += p;
      }
      li[r] = li[r] * a + ssum;
      mi[r] = newm;
      al[r] = a;
    }
    __syncthreads();

    // rescale O, then O += P @ V
    {
      int hi = (lane >> 4) & 1;
      for (int tt = 0; tt < 2; ++tt) {
        int t = t0 + tt, ti = t >> 2, tj = t & 3;
        f32x8& O = tt ? O1 : O0;
#pragma unroll
        for (int r = 0; r < 8; ++r) O[r] *= al[ti * 16 + hi * 8 + r];
        for (int kk = 0; kk < 64; kk += 32) {
          bf16x16 a = frag_a_w(Pb, ti * 16, kk, lane);
          bf16x16 b = frag_bt_w(Vt, kk, tj * 16, lane);  // B[key][d] = Vt[d][key]
          O = wmma_bf16(a, b, O);
        }
      }
    }
    __syncthreads();
  }

  int hi = (lane >> 4) & 1, nn = lane & 15;
  for (int tt = 0; tt < 2; ++tt) {
    int t = t0 + tt, ti = t >> 2, tj = t & 3;
    f32x8& O = tt ? O1 : O0;
#pragma unroll
    for (int r = 0; r < 8; ++r) {
      int lr = ti * 16 + hi * 8 + r;
      out[((size_t)(h * LN) + mblk * 64 + lr) * DN + tj * 16 + nn] = O[r] / li[lr];
    }
  }
}

// ---------------- kernel 5: linear attention + projection, adds into out ----------------
__global__ void __launch_bounds__(256)
linear_kernel(const float* __restrict__ q, const float* __restrict__ kvsum,
              const float* __restrict__ ksum, const float* __restrict__ Wp,
              const float* __restrict__ bp, float* __restrict__ out) {
  __shared__ __bf16 QFb[64 * LDB];  // q_feat, row-major [l][d]
  __shared__ __bf16 KVt[64 * LDB];  // kv_sum transposed [e][d]
  __shared__ __bf16 OL[64 * LDB];   // o_l, row-major [l][d]
  __shared__ __bf16 Wb[64 * LDB];   // W row-major [e][d] (B=W^T reads rows)
  __shared__ float  DEN[64];
  int hb = blockIdx.x;
  int h = hb >> 5, cblk = hb & 31;
  int tid = threadIdx.x, lane = tid & 31, wid = tid >> 5;
  int t0 = wid * 2;

  stage_transposed(kvsum + (size_t)h * 4096, KVt, tid);  // KVt[e][d] = kv[d][e]
  stage_rowmajor(Wp, Wb, tid, 1.0f);
  const float* qbase = q + ((size_t)(h * LN) + cblk * 64) * DN;
  if (tid < 64) {  // q_feat softmax + denominator (f32)
    int r = tid;
    const float* row = qbase + r * 64;
    float mx = -INFINITY;
    for (int d = 0; d < 64; ++d) mx = fmaxf(mx, row[d]);
    float sum = 0.f;
    for (int d = 0; d < 64; ++d) sum += __expf(row[d] - mx);
    float inv = 1.0f / sum;
    float den = 0.f;
    for (int d = 0; d < 64; ++d) {
      float f = __expf(row[d] - mx) * inv;
      QFb[r * LDB + d] = (__bf16)f;
      den += f * ksum[h * 64 + d];
    }
    DEN[r] = den + 1e-5f;
  }
  __syncthreads();

  // num = q_feat @ kv_sum
  f32x8 N0 = {}, N1 = {};
  for (int tt = 0; tt < 2; ++tt) {
    int t = t0 + tt, ti = t >> 2, tj = t & 3;
    f32x8& Nv = tt ? N1 : N0;
    for (int kk = 0; kk < 64; kk += 32) {
      bf16x16 a = frag_a_w(QFb, ti * 16, kk, lane);
      bf16x16 b = frag_bt_w(KVt, kk, tj * 16, lane);  // B[d][e] = KVt[e][d]
      Nv = wmma_bf16(a, b, Nv);
    }
  }
  int hi = (lane >> 4) & 1, nn = lane & 15;
  for (int tt = 0; tt < 2; ++tt) {  // o_l = num / den -> bf16 LDS
    int t = t0 + tt, ti = t >> 2, tj = t & 3;
    f32x8& Nv = tt ? N1 : N0;
#pragma unroll
    for (int r = 0; r < 8; ++r) {
      int lr = ti * 16 + hi * 8 + r;
      OL[lr * LDB + tj * 16 + nn] = (__bf16)(Nv[r] / DEN[lr]);
    }
  }
  __syncthreads();

  // proj = o_l @ W^T ; out += proj + b
  for (int tt = 0; tt < 2; ++tt) {
    int t = t0 + tt, ti = t >> 2, tj = t & 3;
    f32x8 Pv = {};
    for (int kk = 0; kk < 64; kk += 32) {
      bf16x16 a = frag_a_w(OL, ti * 16, kk, lane);
      bf16x16 b = frag_bt_w(Wb, kk, tj * 16, lane);  // B[d][e] = W[e][d]
      Pv = wmma_bf16(a, b, Pv);
    }
#pragma unroll
    for (int r = 0; r < 8; ++r) {
      int lr = ti * 16 + hi * 8 + r;
      size_t oi = ((size_t)(h * LN) + cblk * 64 + lr) * DN + tj * 16 + nn;
      out[oi] += Pv[r] + bp[tj * 16 + nn];
    }
  }
}

// ---------------- launch ----------------
extern "C" void kernel_launch(void* const* d_in, const int* in_sizes, int n_in,
                              void* d_out, int out_size, void* d_ws, size_t ws_size,
                              hipStream_t stream) {
  const float* q  = (const float*)d_in[0];
  const float* k  = (const float*)d_in[1];
  const float* v  = (const float*)d_in[2];
  const float* Wp = (const float*)d_in[3];
  const float* bp = (const float*)d_in[4];
  float* out = (float*)d_out;

  // workspace layout (floats): qm | km | lut(int) | kvsum | ksum (~545 KB)
  float* qm    = (float*)d_ws;
  float* km    = qm + HN * MB * DN;
  int*   lut   = (int*)(km + HN * NB * DN);
  float* kvsum = (float*)(lut + HN * MB * 8);
  float* ksum  = kvsum + HN * DN * DN;

  means_kernel <<<HN * MB, 128, 0, stream>>>(q, k, qm, km);
  topk_kernel  <<<HN * MB,  32, 0, stream>>>(qm, km, lut);
  kv_kernel    <<<HN,      256, 0, stream>>>(k, v, kvsum, ksum);
  attn_kernel  <<<HN * MB, 256, 0, stream>>>(q, k, v, lut, out);
  linear_kernel<<<HN * MB, 256, 0, stream>>>(q, kvsum, ksum, Wp, bp, out);
}
